// Decoder_83571473645992
// MI455X (gfx1250) — matrix operand order repaired and checked by reading, not compiled, not run on hardware
//
#include <hip/hip_runtime.h>
#include <cstdint>
#include <cstddef>

// Problem constants (match reference)
#define B_    4
#define TTOK  512
#define LIMG  256
#define TSEQ  768          // LIMG + TTOK
#define CD    768
#define HH    12
#define DH    64
#define NL_   12
#define VV    50257
#define RR    4
#define SCALE_ 8.0f        // ALPHA / R = 32 / 4

typedef __attribute__((ext_vector_type(16))) _Float16 v16h;
typedef __attribute__((ext_vector_type(8)))  float    v8f;
typedef __attribute__((ext_vector_type(2)))  __fp16   fp16x2;   // cvt_pkrtz result type

union Frag16 { v16h v; unsigned u[8]; };

__device__ inline unsigned short f2h_bits(float f) {
    union { _Float16 h; unsigned short s; } cv;
    cv.h = (_Float16)f;
    return cv.s;
}

// pack two fp32 -> two f16 in one VGPR (v_cvt_pk_rtz_f16_f32)
__device__ inline unsigned pk2h(float a, float b) {
    union { fp16x2 h; unsigned u; } cv;
    cv.h = __builtin_amdgcn_cvt_pkrtz(a, b);
    return cv.u;
}

__device__ inline float wave_sum(float v) {
    v += __shfl_xor(v, 16, 32);
    v += __shfl_xor(v, 8, 32);
    v += __shfl_xor(v, 4, 32);
    v += __shfl_xor(v, 2, 32);
    v += __shfl_xor(v, 1, 32);
    return v;
}

// ---------------------------------------------------------------------------
// Generic GEMM: Out[M,N] = A[M,K] @ W + bias  (fp32 in/out, f16 WMMA core)
// transB==0: W is [K,N] row-major, N % 128 == 0 at all call sites.
// transB==1: W is [N,K] row-major (wte), N guarded (uniform per thread).
// Requires M % 128 == 0, K % 32 == 0 (true for all call sites).
// Workgroup 256 thr = 8 waves (4 M x 2 N); wave tile 32x64 = 2x4 WMMA tiles.
// LDS stride 40 halves = 80 B: 16B-aligned vector stores, bank step 20 -> no
// conflicts on the 16-lane fragment reads.
// ---------------------------------------------------------------------------
#define LDSW 40

__global__ __launch_bounds__(256) void gemm_f16_wmma(
    const float* __restrict__ A, const float* __restrict__ W,
    const float* __restrict__ bias, float* __restrict__ Out,
    int M, int N, int K, int transB)
{
    __shared__ __align__(16) unsigned short As[128 * LDSW];
    __shared__ __align__(16) unsigned short Bs[128 * LDSW];

    const int tid  = threadIdx.x;
    const int lane = tid & 31;
    const int wave = tid >> 5;
    const int wm   = wave >> 1;          // 0..3  (M direction)
    const int wn   = wave & 1;           // 0..1  (N direction)
    const int tileM = blockIdx.y * 128;
    const int tileN = blockIdx.x * 128;

    // staging coordinates (A-style: 128 rows x 8 float4; B-trans same shape)
    const int sm = tid >> 3;             // 0..31  (+32 per i)
    const int sk = (tid & 7) * 4;        // 0,4,..,28
    // B non-trans staging coordinates
    const int bk = tid >> 5;             // 0..7   (+8 per i)
    const int bnn = (tid & 31) * 4;      // 0,4,..,124

    v8f acc[2][4];
#pragma unroll
    for (int i = 0; i < 2; ++i)
#pragma unroll
        for (int j = 0; j < 4; ++j)
#pragma unroll
            for (int r = 0; r < 8; ++r) acc[i][j][r] = 0.0f;

    for (int kk = 0; kk < K; kk += 32) {
        // ---- speculative prefetch of the next K-tile (global_prefetch_b8) --
        int kk2 = kk + 32;
        if (kk2 < K) {
            if (tid < 128) {
                // A tile row = exactly one 128B line
                __builtin_prefetch(A + (size_t)(tileM + tid) * K + kk2, 0, 3);
            } else if (!transB) {
                // 32 rows x 512B -> 128 line prefetches
                int k = (tid - 128) >> 2;
                int seg = (tid - 128) & 3;
                __builtin_prefetch(W + (size_t)(kk2 + k) * N + tileN + seg * 32, 0, 3);
            } else {
                int n = tileN + (tid - 128);
                if (n < N)
                    __builtin_prefetch(W + (size_t)n * K + kk2, 0, 3);
            }
        }

        // ================= phase 1: issue ALL global loads =================
        float4 fa[4];
#pragma unroll
        for (int i = 0; i < 4; ++i)
            fa[i] = *(const float4*)(A + (size_t)(tileM + sm + i * 32) * K + kk + sk);

        float4 fb[4];
        if (!transB) {
#pragma unroll
            for (int i = 0; i < 4; ++i)
                fb[i] = *(const float4*)(W + (size_t)(kk + bk + i * 8) * N + tileN + bnn);
        } else {
#pragma unroll
            for (int i = 0; i < 4; ++i) {
                int gn = tileN + sm + i * 32;
                fb[i] = make_float4(0.f, 0.f, 0.f, 0.f);
                if (gn < N) fb[i] = *(const float4*)(W + (size_t)gn * K + kk + sk);
            }
        }

        // ================= phase 2: convert + store to LDS =================
#pragma unroll
        for (int i = 0; i < 4; ++i) {
            unsigned* dst = (unsigned*)&As[(sm + i * 32) * LDSW + sk];
            dst[0] = pk2h(fa[i].x, fa[i].y);
            dst[1] = pk2h(fa[i].z, fa[i].w);
        }
        if (!transB) {
#pragma unroll
            for (int i = 0; i < 4; ++i) {
                int k = bk + i * 8;
                Bs[(bnn + 0) * LDSW + k] = f2h_bits(fb[i].x);
                Bs[(bnn + 1) * LDSW + k] = f2h_bits(fb[i].y);
                Bs[(bnn + 2) * LDSW + k] = f2h_bits(fb[i].z);
                Bs[(bnn + 3) * LDSW + k] = f2h_bits(fb[i].w);
            }
        } else {
#pragma unroll
            for (int i = 0; i < 4; ++i) {
                unsigned* dst = (unsigned*)&Bs[(sm + i * 32) * LDSW + sk];
                dst[0] = pk2h(fb[i].x, fb[i].y);
                dst[1] = pk2h(fb[i].z, fb[i].w);
            }
        }
        __syncthreads();

        // ---- build wave32 fragments per CDNA5 16x16x32 layouts ----
        v16h afr[2], bfr[4];
        {
            // A 16x32: lanes 0-15 row M=lane, K base 0; lanes 16-31 K base 8.
            // VGPR e pairs: k = kbase + (e<8 ? e : e+8)
            int am = lane & 15, akb = (lane >> 4) * 8;
#pragma unroll
            for (int mi = 0; mi < 2; ++mi) {
                int row = wm * 32 + mi * 16 + am;
                Frag16 fra;
#pragma unroll
                for (int j = 0; j < 8; ++j) {
                    int k0 = akb + ((j < 4) ? (2 * j) : (2 * j + 8));
                    fra.u[j] = *(const unsigned*)(&As[row * LDSW + k0]);
                }
                afr[mi] = fra.v;
            }
            // B 32x16: lanes 0-15 col N=lane, K base 0; lanes 16-31 K base 8 (same order as A).
            int bn = lane & 15, bkb = (lane >> 4) * 8;
#pragma unroll
            for (int ni = 0; ni < 4; ++ni) {
                int col = wn * 64 + ni * 16 + bn;
                Frag16 frb;
#pragma unroll
                for (int j = 0; j < 8; ++j)
                    frb.u[j] = *(const unsigned*)(&Bs[col * LDSW + bkb + ((j < 4) ? (2 * j) : (2 * j + 8))]);
                bfr[ni] = frb.v;
            }
        }
#pragma unroll
        for (int mi = 0; mi < 2; ++mi)
#pragma unroll
            for (int ni = 0; ni < 4; ++ni)
                acc[mi][ni] = __builtin_amdgcn_wmma_f32_16x16x32_f16(
                    false, afr[mi], false, bfr[ni], (short)0, acc[mi][ni],
                    false, false);
        __syncthreads();
    }

    // ---- epilogue: C/D layout -> global. lane<16: M=r, N=lane; else M=r+8 ----
    int cn = lane & 15;
    int mo = (lane >> 4) * 8;
#pragma unroll
    for (int mi = 0; mi < 2; ++mi)
#pragma unroll
        for (int ni = 0; ni < 4; ++ni) {
            int gn = tileN + wn * 64 + ni * 16 + cn;
            if (gn < N) {
                float bv = bias ? bias[gn] : 0.0f;
#pragma unroll
                for (int r = 0; r < 8; ++r) {
                    int gm = tileM + wm * 32 + mi * 16 + mo + r;
                    Out[(size_t)gm * N + gn] = acc[mi][ni][r] + bv;
                }
            }
        }
}

// ---------------------------------------------------------------------------
// LayerNorm: one 256-thread block per row of width CD.
// ---------------------------------------------------------------------------
__global__ __launch_bounds__(256) void layernorm_k(
    const float* __restrict__ x, const float* __restrict__ g,
    const float* __restrict__ b, float* __restrict__ out)
{
    __shared__ float red[8];
    const int row = blockIdx.x;
    const float* xr = x + (size_t)row * CD;
    const int tid = threadIdx.x, lane = tid & 31, wv = tid >> 5;

    float s = 0.0f;
    for (int c = tid; c < CD; c += 256) s += xr[c];
    s = wave_sum(s);
    if (lane == 0) red[wv] = s;
    __syncthreads();
    float mean = (red[0] + red[1] + red[2] + red[3] +
                  red[4] + red[5] + red[6] + red[7]) * (1.0f / CD);
    __syncthreads();

    float v = 0.0f;
    for (int c = tid; c < CD; c += 256) { float d = xr[c] - mean; v += d * d; }
    v = wave_sum(v);
    if (lane == 0) red[wv] = v;
    __syncthreads();
    float var = (red[0] + red[1] + red[2] + red[3] +
                 red[4] + red[5] + red[6] + red[7]) * (1.0f / CD);
    float rstd = rsqrtf(var + 1e-5f);

    float* orow = out + (size_t)row * CD;
    for (int c = tid; c < CD; c += 256)
        orow[c] = (xr[c] - mean) * rstd * g[c] + b[c];
}

// ---------------------------------------------------------------------------
// Embedding: x[b,t,:] = img for t<LIMG else wte[token] + wpe[t-LIMG]
// ---------------------------------------------------------------------------
__global__ void embed_k(const int* __restrict__ token,
                        const float* __restrict__ img,
                        const float* __restrict__ wte,
                        const float* __restrict__ wpe,
                        float* __restrict__ x)
{
    size_t i = (size_t)blockIdx.x * 256 + threadIdx.x;
    if (i >= (size_t)B_ * TSEQ * CD) return;
    int c  = (int)(i % CD);
    int bt = (int)(i / CD);
    int b = bt / TSEQ, t = bt % TSEQ;
    if (t < LIMG) {
        x[i] = img[((size_t)(b * LIMG + t)) * CD + c];
    } else {
        int tt = t - LIMG;
        int tk = token[b * TTOK + tt];
        x[i] = wte[(size_t)tk * CD + c] + wpe[(size_t)tt * CD + c];
    }
}

// ---------------------------------------------------------------------------
// LoRA stage 1: tmp[M,4] = in[M,K] @ Am[K,4]
// ---------------------------------------------------------------------------
__global__ void lora_ar_k(const float* __restrict__ in,
                          const float* __restrict__ Am,
                          float* __restrict__ tmp, int M, int K)
{
    int t = blockIdx.x * 256 + threadIdx.x;
    if (t >= M * RR) return;
    int m = t >> 2, r = t & 3;
    const float* ir = in + (size_t)m * K;
    float s = 0.0f;
    for (int k = 0; k < K; ++k) s += ir[k] * Am[k * RR + r];
    tmp[t] = s;
}

// ---------------------------------------------------------------------------
// LoRA stage 2: out[m, n] += SCALE * tmp[m,:] @ Bm[:, n]   (Bm is [4, Ncols])
// ---------------------------------------------------------------------------
__global__ void lora_add_k(const float* __restrict__ tmp,
                           const float* __restrict__ Bm,
                           float* __restrict__ out,
                           int M, int Ncols, int ldOut)
{
    long long t = (long long)blockIdx.x * 256 + threadIdx.x;
    if (t >= (long long)M * Ncols) return;
    int m = (int)(t / Ncols), n = (int)(t % Ncols);
    const float* tr = tmp + (size_t)m * RR;
    float s = tr[0] * Bm[n] + tr[1] * Bm[(size_t)Ncols + n] +
              tr[2] * Bm[2 * (size_t)Ncols + n] + tr[3] * Bm[3 * (size_t)Ncols + n];
    out[(size_t)m * ldOut + n] += SCALE_ * s;
}

// ---------------------------------------------------------------------------
// Attention: wave per (b, h, q-row), online softmax over keys 0..qi.
// qkv is [B*T, 3C]: q at +0, k at +CD, v at +2*CD within a row.
// If att != nullptr (layer 11) also emits exact normalized probabilities.
// ---------------------------------------------------------------------------
__global__ __launch_bounds__(256) void attn_k(const float* __restrict__ qkv,
                                              float* __restrict__ y,
                                              float* __restrict__ att)
{
    const int wid  = blockIdx.x * 8 + (threadIdx.x >> 5);
    const int lane = threadIdx.x & 31;
    if (wid >= B_ * HH * TSEQ) return;
    const int b   = wid / (HH * TSEQ);
    const int rem = wid % (HH * TSEQ);
    const int h   = rem / TSEQ;
    const int qi  = rem % TSEQ;
    const int LDQ = 3 * CD;

    const float* qp = qkv + (size_t)(b * TSEQ + qi) * LDQ + h * DH;
    const float q0 = qp[lane], q1 = qp[lane + 32];

    float mx = -3.0e38f, sum = 0.0f, a0 = 0.0f, a1 = 0.0f;
    for (int ki = 0; ki <= qi; ++ki) {
        const float* kp = qkv + (size_t)(b * TSEQ + ki) * LDQ + CD + h * DH;
        float s = q0 * kp[lane] + q1 * kp[lane + 32];
        s = wave_sum(s) * 0.125f;              // 1/sqrt(64)
        float nm   = fmaxf(mx, s);
        float corr = __expf(mx - nm);
        float p    = __expf(s - nm);
        const float* vp = kp + CD;
        sum = sum * corr + p;
        a0  = a0 * corr + p * vp[lane];
        a1  = a1 * corr + p * vp[lane + 32];
        mx  = nm;
    }
    const float inv = 1.0f / sum;
    float* yp = y + (size_t)(b * TSEQ + qi) * CD + h * DH;
    yp[lane]      = a0 * inv;
    yp[lane + 32] = a1 * inv;

    if (att) {
        float* ar = att + ((size_t)(b * HH + h) * TSEQ + qi) * TSEQ;
        for (int ki = 0; ki <= qi; ++ki) {
            const float* kp = qkv + (size_t)(b * TSEQ + ki) * LDQ + CD + h * DH;
            float s = q0 * kp[lane] + q1 * kp[lane + 32];
            s = wave_sum(s) * 0.125f;
            if (lane == 0) ar[ki] = __expf(s - mx) * inv;
        }
        for (int ki = qi + 1 + lane; ki < TSEQ; ki += 32) ar[ki] = 0.0f;
    }
}

// ---------------------------------------------------------------------------
// Elementwise helpers
// ---------------------------------------------------------------------------
__global__ void gelu_k(float* __restrict__ x, long long n)
{
    long long i = (long long)blockIdx.x * 256 + threadIdx.x;
    if (i >= n) return;
    float v = x[i];
    float c = v + 0.044715f * v * v * v;
    x[i] = 0.5f * v * (1.0f + tanhf(0.7978845608028654f * c));
}

__global__ void add_k(float* __restrict__ x, const float* __restrict__ t, long long n)
{
    long long i = (long long)blockIdx.x * 256 + threadIdx.x;
    if (i < n) x[i] += t[i];
}

__global__ void copyxt_k(const float* __restrict__ h, float* __restrict__ xt)
{
    size_t i = (size_t)blockIdx.x * 256 + threadIdx.x;
    if (i >= (size_t)B_ * TTOK * CD) return;
    int c   = (int)(i % CD);
    int row = (int)(i / CD);
    int b = row / TTOK, t = row % TTOK;
    xt[i] = h[((size_t)(b * TSEQ + LIMG + t)) * CD + c];
}

// ---------------------------------------------------------------------------
// Host driver
// ---------------------------------------------------------------------------
static inline unsigned cdiv(long long a, long long b) { return (unsigned)((a + b - 1) / b); }

extern "C" void kernel_launch(void* const* d_in, const int* in_sizes, int n_in,
                              void* d_out, int out_size, void* d_ws, size_t ws_size,
                              hipStream_t stream)
{
    (void)in_sizes; (void)n_in; (void)out_size; (void)ws_size;

    const int*   token = (const int*)  d_in[0];
    const float* img   = (const float*)d_in[1];
    const float* wte   = (const float*)d_in[2];
    const float* wpe   = (const float*)d_in[3];
    const float* ln1g  = (const float*)d_in[4];
    const float* ln1b  = (const float*)d_in[5];
    const float* Wqkv  = (const float*)d_in[6];
    const float* bqkv  = (const float*)d_in[7];
    const float* Aq    = (const float*)d_in[8];
    const float* Bq    = (const float*)d_in[9];
    const float* Av    = (const float*)d_in[10];
    const float* Bv    = (const float*)d_in[11];
    const float* Wo    = (const float*)d_in[12];
    const float* bo    = (const float*)d_in[13];
    const float* Ao    = (const float*)d_in[14];
    const float* Bo    = (const float*)d_in[15];
    const float* ln2g  = (const float*)d_in[16];
    const float* ln2b  = (const float*)d_in[17];
    const float* Wfc   = (const float*)d_in[18];
    const float* bfc   = (const float*)d_in[19];
    const float* Afc   = (const float*)d_in[20];
    const float* Bfc   = (const float*)d_in[21];
    const float* Wp2   = (const float*)d_in[22];
    const float* bp2   = (const float*)d_in[23];
    const float* Ap2   = (const float*)d_in[24];
    const float* Bp2   = (const float*)d_in[25];
    const float* lnfg  = (const float*)d_in[26];
    const float* lnfb  = (const float*)d_in[27];
    const float* Alm   = (const float*)d_in[28];
    const float* Blm   = (const float*)d_in[29];

    float* out    = (float*)d_out;
    float* logits = out;                                    // [B, TTOK, V]
    float* attmap = out + (size_t)B_ * TTOK * VV;           // [B, H, T, T]

    const size_t BT = (size_t)B_ * TSEQ;                    // 3072
    float* ws  = (float*)d_ws;
    float* x   = ws;  ws += BT * CD;                        // residual stream
    float* h   = ws;  ws += BT * CD;                        // LN output
    float* qkv = ws;  ws += BT * 3 * CD;                    // fused qkv
    float* yb  = ws;  ws += BT * CD;                        // attention output
    float* tb  = ws;  ws += BT * CD;                        // projection temp
    float* mb  = ws;  ws += BT * 4 * CD;                    // MLP hidden
    float* lt  = ws;  ws += BT * RR;                        // LoRA rank temp
    float* xt  = ws;  ws += (size_t)B_ * TTOK * CD;         // final token slice

    const dim3 blk(256);
    const long long nXC = (long long)BT * CD;               // 2.36M

    embed_k<<<cdiv(nXC, 256), blk, 0, stream>>>(token, img, wte, wpe, x);

    for (int i = 0; i < NL_; ++i) {
        // ---- attention block ----
        layernorm_k<<<(unsigned)BT, blk, 0, stream>>>(x, ln1g + (size_t)i * CD, ln1b + (size_t)i * CD, h);
        gemm_f16_wmma<<<dim3(18, 24), blk, 0, stream>>>(h, Wqkv + (size_t)i * CD * 3 * CD,
                                                        bqkv + (size_t)i * 3 * CD, qkv,
                                                        3072, 3 * CD, CD, 0);
        lora_ar_k<<<cdiv(3072 * RR, 256), blk, 0, stream>>>(h, Aq + (size_t)i * CD * RR, lt, 3072, CD);
        lora_add_k<<<cdiv((long long)3072 * CD, 256), blk, 0, stream>>>(lt, Bq + (size_t)i * RR * CD, qkv, 3072, CD, 3 * CD);
        lora_ar_k<<<cdiv(3072 * RR, 256), blk, 0, stream>>>(h, Av + (size_t)i * CD * RR, lt, 3072, CD);
        lora_add_k<<<cdiv((long long)3072 * CD, 256), blk, 0, stream>>>(lt, Bv + (size_t)i * RR * CD, qkv + 2 * CD, 3072, CD, 3 * CD);

        attn_k<<<(B_ * HH * TSEQ) / 8, blk, 0, stream>>>(qkv, yb, (i == NL_ - 1) ? attmap : nullptr);

        gemm_f16_wmma<<<dim3(6, 24), blk, 0, stream>>>(yb, Wo + (size_t)i * CD * CD,
                                                       bo + (size_t)i * CD, tb, 3072, CD, CD, 0);
        lora_ar_k<<<cdiv(3072 * RR, 256), blk, 0, stream>>>(yb, Ao + (size_t)i * CD * RR, lt, 3072, CD);
        lora_add_k<<<cdiv((long long)3072 * CD, 256), blk, 0, stream>>>(lt, Bo + (size_t)i * RR * CD, tb, 3072, CD, CD);
        add_k<<<cdiv(nXC, 256), blk, 0, stream>>>(x, tb, nXC);

        // ---- MLP block ----
        layernorm_k<<<(unsigned)BT, blk, 0, stream>>>(x, ln2g + (size_t)i * CD, ln2b + (size_t)i * CD, h);
        gemm_f16_wmma<<<dim3(24, 24), blk, 0, stream>>>(h, Wfc + (size_t)i * CD * 4 * CD,
                                                        bfc + (size_t)i * 4 * CD, mb,
                                                        3072, 4 * CD, CD, 0);
        lora_ar_k<<<cdiv(3072 * RR, 256), blk, 0, stream>>>(h, Afc + (size_t)i * CD * RR, lt, 3072, CD);
        lora_add_k<<<cdiv((long long)3072 * 4 * CD, 256), blk, 0, stream>>>(lt, Bfc + (size_t)i * RR * 4 * CD, mb, 3072, 4 * CD, 4 * CD);
        gelu_k<<<cdiv((long long)BT * 4 * CD, 256), blk, 0, stream>>>(mb, (long long)BT * 4 * CD);

        gemm_f16_wmma<<<dim3(6, 24), blk, 0, stream>>>(mb, Wp2 + (size_t)i * 4 * CD * CD,
                                                       bp2 + (size_t)i * CD, tb, 3072, CD, 4 * CD, 0);
        lora_ar_k<<<cdiv(3072 * RR, 256), blk, 0, stream>>>(mb, Ap2 + (size_t)i * 4 * CD * RR, lt, 3072, 4 * CD);
        lora_add_k<<<cdiv((long long)3072 * CD, 256), blk, 0, stream>>>(lt, Bp2 + (size_t)i * RR * CD, tb, 3072, CD, CD);
        add_k<<<cdiv(nXC, 256), blk, 0, stream>>>(x, tb, nXC);
    }

    // ---- LM head ----
    layernorm_k<<<(unsigned)BT, blk, 0, stream>>>(x, lnfg, lnfb, h);
    copyxt_k<<<cdiv((long long)B_ * TTOK * CD, 256), blk, 0, stream>>>(h, xt);
    gemm_f16_wmma<<<dim3((VV + 127) / 128, (B_ * TTOK) / 128), blk, 0, stream>>>(
        xt, wte, nullptr, logits, B_ * TTOK, VV, CD, 1);
    lora_ar_k<<<cdiv((long long)B_ * TTOK * RR, 256), blk, 0, stream>>>(xt, Alm, lt, B_ * TTOK, CD);
    lora_add_k<<<cdiv((long long)B_ * TTOK * VV, 256), blk, 0, stream>>>(lt, Blm, logits, B_ * TTOK, VV, VV);
}
